// LSTM_DECODER_91044716741128
// MI455X (gfx1250) — matrix operand order repaired and checked by reading, not compiled, not run on hardware
//
#include <hip/hip_runtime.h>
#include <hip/hip_bf16.h>

typedef __attribute__((ext_vector_type(16))) __bf16 v16bf;
typedef __attribute__((ext_vector_type(8)))  float  v8f;

#define NENV   4096
#define SEQ    128
#define LATENT 64
#define EMB    256
#define HID    1024
#define ROWS   32   // envs per workgroup (2 row tiles of 16)

#define WMMA_BF16 __builtin_amdgcn_wmma_f32_16x16x32_bf16

// ---------------------------------------------------------------------------
// Pre-pass: convert an fp32 weight matrix W[Ktot x Ntot] into bf16 fragments
// in WMMA B-operand layout.  Fragment grid: KC = Ktot/32, NC = Ntot/16.
// lane = N%16 + 16*half ; K = kc*32 + (e/8)*16 + half*8 + e%8
// Each lane's 16 bf16 (32B) are contiguous -> 2x global_load_b128 per frag.
// ---------------------------------------------------------------------------
__global__ void convert_swizzle(const float* __restrict__ src,
                                __bf16* __restrict__ dst,
                                int KC, int NC, int Ntot) {
    int t = blockIdx.x * 256 + threadIdx.x;
    int total = KC * NC * 512;
    if (t >= total) return;
    int e    = t & 15;
    int lane = (t >> 4) & 31;
    int fc   = t >> 9;          // fragment index kc*NC + nc
    int nc   = fc % NC;
    int kc   = fc / NC;
    int N    = nc * 16 + (lane & 15);
    int K    = kc * 32 + (e >> 3) * 16 + (lane >> 4) * 8 + (e & 7);
    dst[t] = (__bf16)src[(size_t)K * Ntot + N];
}

// ---------------------------------------------------------------------------
// A-fragment (16x32 bf16) index helper: element (row M<16, col K<32)
//   half = (K/8)%2 ; lane = M + 16*half ; e = (K/16)*8 + (K%8)
// ---------------------------------------------------------------------------
__device__ __forceinline__ int afrag_idx(int kc, int M, int ko) {
    int hf = (ko >> 3) & 1;
    int e  = ((ko >> 4) << 3) + (ko & 7);
    int l2 = M + (hf << 4);
    return ((kc * 32 + l2) << 4) + e;
}

__device__ __forceinline__ float sigm(float x) {
    return 1.0f / (1.0f + __expf(-x));
}

// ---------------------------------------------------------------------------
// Persistent GRU decoder: one workgroup = 32 envs (2 row tiles), 128 steps.
// 256 threads = 8 waves (wave32).  GRU and w3 matmuls via v_wmma_f32_16x16x32_bf16.
// gi and gh accumulate into the SAME registers for the r/z gates (the GRU
// only ever uses their sum), so a wave job needs just 8 accumulators.
// Weight fragments are addressed as uniform-base + 32-bit lane offset +
// compile-time immediate (kc stride 192KB, gate stride 64KB, max 6.2MB
// < signed-24-bit) so no 64-bit per-job address pairs stay live.
// ---------------------------------------------------------------------------
__global__ __launch_bounds__(256, 1) void gru_decoder_kernel(
    const float* __restrict__ z,    const float* __restrict__ h0,
    const float* __restrict__ w1,   const float* __restrict__ b1,
    const float* __restrict__ w2,   const float* __restrict__ b2,
    const float* __restrict__ bih,  const float* __restrict__ bhh,
    const float* __restrict__ b3,   const float* __restrict__ w4,
    const float* __restrict__ b4,
    const char* __restrict__ wihq,  // 8  kc x 192 nc (bf16 frags, 1KB/tile)
    const char* __restrict__ whhq,  // 32 kc x 192 nc
    const char* __restrict__ w3q,   // 32 kc x 32  nc
    float* __restrict__ out, float* __restrict__ hout)
{
    __shared__ v16bf h_frag[2][2][32 * 32];  // [buf][rt][kc*32+lane] bf16 A layout
    __shared__ v16bf x_frag[2][8 * 32];      // [rt][kc*32+lane]
    __shared__ float o1_lds[ROWS * 512];     // relu(h@w3)
    __shared__ float x1_lds[ROWS * 32];      // relu(in@w1)
    __shared__ float state_lds[ROWS * 2];    // fed-back output
    __shared__ float red_lds[ROWS * 2 * 4];  // final-GEMM partials

    const int tid  = threadIdx.x;
    const int lane = tid & 31;
    const int wave = tid >> 5;
    const int env0 = blockIdx.x * ROWS;

    // ---- init: state = -2, h_frag[0] = bf16(h0) in A layout ----
    for (int i = tid; i < ROWS * 2; i += 256) state_lds[i] = -2.0f;
    for (int i = tid; i < ROWS * HID; i += 256) {
        int r = i / HID, c = i % HID;
        __bf16* hf = (__bf16*)&h_frag[0][r >> 4][0];
        hf[afrag_idx(c >> 5, r & 15, c & 31)] = (__bf16)h0[(size_t)(env0 + r) * HID + c];
    }
    __syncthreads();

    int cur = 0;
    for (int t = 0; t < SEQ; ++t) {
        const int nxt = cur ^ 1;

        // ---- stage 0a: x1 = relu([state, z_t] @ w1 + b1)  (32x66 @ 66x32) ----
        for (int i = tid; i < ROWS * 32; i += 256) {
            int r = i >> 5, c = i & 31;
            float acc = b1[c];
            acc += state_lds[r * 2 + 0] * w1[0 * 32 + c];
            acc += state_lds[r * 2 + 1] * w1[1 * 32 + c];
            const float* zr = z + ((size_t)(env0 + r) * SEQ + t) * LATENT;
            #pragma unroll 8
            for (int k = 0; k < LATENT; ++k)
                acc += zr[k] * w1[(2 + k) * 32 + c];
            x1_lds[i] = fmaxf(acc, 0.0f);
        }
        __syncthreads();

        // ---- stage 0b: x = relu(x1 @ w2 + b2) -> x_frag (A layout bf16) ----
        for (int i = tid; i < ROWS * EMB; i += 256) {
            int r = i >> 8, c = i & 255;
            float acc = b2[c];
            #pragma unroll 8
            for (int k = 0; k < 32; ++k)
                acc += x1_lds[r * 32 + k] * w2[k * EMB + c];
            __bf16* xf = (__bf16*)&x_frag[r >> 4][0];
            xf[afrag_idx(c >> 5, r & 15, c & 31)] = (__bf16)fmaxf(acc, 0.0f);
        }
        __syncthreads();

        // ---- GRU: (x@wih + h@whh) fused accumulation + gate math ----
        // 64 hidden-col tiles; r/z/n gate tiles at ct, ct+64, ct+128.
        for (int ct = wave; ct < 64; ct += 8) {
            v8f ar0 = {}, az0 = {}, in0 = {}, hn0 = {};
            v8f ar1 = {}, az1 = {}, in1 = {}, hn1 = {};
            const unsigned joff = (unsigned)ct * 1024u + (unsigned)lane * 32u;

            // gi: K = EMB = 256 -> 8 chunks (r/z/i_n paths)
            {
                const char* wp = wihq + joff;
                #pragma unroll
                for (int kc = 0; kc < 8; ++kc) {
                    v16bf a0 = x_frag[0][kc * 32 + lane];
                    v16bf a1 = x_frag[1][kc * 32 + lane];
                    v16bf br = *(const v16bf*)(wp + kc * 196608);
                    v16bf bz = *(const v16bf*)(wp + kc * 196608 + 65536);
                    v16bf bn = *(const v16bf*)(wp + kc * 196608 + 131072);
                    ar0 = WMMA_BF16(false, a0, false, br, (short)0, ar0, false, false);
                    ar1 = WMMA_BF16(false, a1, false, br, (short)0, ar1, false, false);
                    az0 = WMMA_BF16(false, a0, false, bz, (short)0, az0, false, false);
                    az1 = WMMA_BF16(false, a1, false, bz, (short)0, az1, false, false);
                    in0 = WMMA_BF16(false, a0, false, bn, (short)0, in0, false, false);
                    in1 = WMMA_BF16(false, a1, false, bn, (short)0, in1, false, false);
                }
            }
            // gh: K = HID = 1024 -> 32 chunks; r/z continue in the same
            // accumulators, h_n goes to its own (needed for r * h_n).
            {
                const char* wp = whhq + joff;
                #pragma unroll
                for (int kc = 0; kc < 32; ++kc) {
                    v16bf a0 = h_frag[cur][0][kc * 32 + lane];
                    v16bf a1 = h_frag[cur][1][kc * 32 + lane];
                    v16bf br = *(const v16bf*)(wp + kc * 196608);
                    v16bf bz = *(const v16bf*)(wp + kc * 196608 + 65536);
                    v16bf bn = *(const v16bf*)(wp + kc * 196608 + 131072);
                    ar0 = WMMA_BF16(false, a0, false, br, (short)0, ar0, false, false);
                    ar1 = WMMA_BF16(false, a1, false, br, (short)0, ar1, false, false);
                    az0 = WMMA_BF16(false, a0, false, bz, (short)0, az0, false, false);
                    az1 = WMMA_BF16(false, a1, false, bz, (short)0, az1, false, false);
                    hn0 = WMMA_BF16(false, a0, false, bn, (short)0, hn0, false, false);
                    hn1 = WMMA_BF16(false, a1, false, bn, (short)0, hn1, false, false);
                }
            }
            // gate math on the two 16x16 accumulator tiles held by this wave.
            // C/D layout: elem g of lane L is (M = g + 8*(L/16), N = L%16)
            const int N  = lane & 15;
            const int ph = lane >> 4;
            const int c  = ct * 16 + N;
            const float BR  = bih[c] + bhh[c];
            const float BZ  = bih[HID + c] + bhh[HID + c];
            const float BIN = bih[2 * HID + c];
            const float BHN = bhh[2 * HID + c];
            const int kc2 = c >> 5, ko2 = c & 31;
            #pragma unroll
            for (int rt = 0; rt < 2; ++rt) {
                const __bf16* hc  = (const __bf16*)&h_frag[cur][rt][0];
                __bf16*       hnb = (__bf16*)&h_frag[nxt][rt][0];
                const v8f& ar = rt ? ar1 : ar0;
                const v8f& az = rt ? az1 : az0;
                const v8f& ain = rt ? in1 : in0;
                const v8f& ahn = rt ? hn1 : hn0;
                #pragma unroll
                for (int g = 0; g < 8; ++g) {
                    int M = g + (ph << 3);
                    float rg = sigm(ar[g] + BR);
                    float ug = sigm(az[g] + BZ);
                    float ng = tanhf(ain[g] + BIN + rg * (ahn[g] + BHN));
                    int idx  = afrag_idx(kc2, M, ko2);
                    float hold = (float)hc[idx];
                    hnb[idx] = (__bf16)((1.0f - ug) * ng + ug * hold);
                }
            }
        }
        __syncthreads();

        // ---- o1 = relu(h_new @ w3 + b3): [32x1024]@[1024x512] ----
        for (int ct = wave; ct < 32; ct += 8) {
            v8f acc0 = {}, acc1 = {};
            const char* wp = w3q + (unsigned)ct * 1024u + (unsigned)lane * 32u;
            #pragma unroll
            for (int kc = 0; kc < 32; ++kc) {
                v16bf a0 = h_frag[nxt][0][kc * 32 + lane];
                v16bf a1 = h_frag[nxt][1][kc * 32 + lane];
                v16bf b  = *(const v16bf*)(wp + kc * 32768);
                acc0 = WMMA_BF16(false, a0, false, b, (short)0, acc0, false, false);
                acc1 = WMMA_BF16(false, a1, false, b, (short)0, acc1, false, false);
            }
            const int N  = lane & 15;
            const int ph = lane >> 4;
            const int c  = ct * 16 + N;
            const float bb = b3[c];
            #pragma unroll
            for (int g = 0; g < 8; ++g) {
                int M = g + (ph << 3);
                o1_lds[(M)      * 512 + c] = fmaxf(acc0[g] + bb, 0.0f);
                o1_lds[(M + 16) * 512 + c] = fmaxf(acc1[g] + bb, 0.0f);
            }
        }
        __syncthreads();

        // ---- o = tanh(o1 @ w4 + b4): 4-way split-K partials + LDS reduce ----
        {
            int o    = tid >> 2;      // output index 0..63  (r*2+cc)
            int part = tid & 3;       // 0..3, each covers 128 k
            int r = o >> 1, cc = o & 1;
            float acc = 0.0f;
            #pragma unroll 8
            for (int k = part * 128; k < part * 128 + 128; ++k)
                acc += o1_lds[r * 512 + k] * w4[k * 2 + cc];
            red_lds[tid] = acc;
        }
        __syncthreads();
        if (tid < ROWS * 2) {
            int r = tid >> 1, cc = tid & 1;
            float acc = b4[cc] + red_lds[tid * 4] + red_lds[tid * 4 + 1]
                               + red_lds[tid * 4 + 2] + red_lds[tid * 4 + 3];
            float o = tanhf(acc);
            out[((size_t)(env0 + r) * SEQ + t) * 2 + cc] = o;
            state_lds[tid] = o;
        }

        // ---- final step: dump new hidden state as fp32 ----
        if (t == SEQ - 1) {
            for (int i = tid; i < ROWS * HID; i += 256) {
                int r = i / HID, c = i % HID;
                const __bf16* hnb = (const __bf16*)&h_frag[nxt][r >> 4][0];
                hout[(size_t)(env0 + r) * HID + c] =
                    (float)hnb[afrag_idx(c >> 5, r & 15, c & 31)];
            }
        }
        __syncthreads();
        cur = nxt;
    }
}

// ---------------------------------------------------------------------------
extern "C" void kernel_launch(void* const* d_in, const int* in_sizes, int n_in,
                              void* d_out, int out_size, void* d_ws, size_t ws_size,
                              hipStream_t stream) {
    const float* z    = (const float*)d_in[0];
    const float* hxs  = (const float*)d_in[1];
    const float* w1   = (const float*)d_in[2];
    const float* b1   = (const float*)d_in[3];
    const float* w2   = (const float*)d_in[4];
    const float* b2   = (const float*)d_in[5];
    const float* wih  = (const float*)d_in[6];
    const float* bih  = (const float*)d_in[7];
    const float* whh  = (const float*)d_in[8];
    const float* bhh  = (const float*)d_in[9];
    const float* w3   = (const float*)d_in[10];
    const float* b3   = (const float*)d_in[11];
    const float* w4   = (const float*)d_in[12];
    const float* b4   = (const float*)d_in[13];

    // Workspace: swizzled bf16 weights
    __bf16* base = (__bf16*)d_ws;
    __bf16* wihq = base;                               //  8*192*512 = 786432
    __bf16* whhq = wihq + 8  * 192 * 512;              // 32*192*512 = 3145728
    __bf16* w3q  = whhq + 32 * 192 * 512;              // 32*32 *512 = 524288

    convert_swizzle<<<(8  * 192 * 512 + 255) / 256, 256, 0, stream>>>(wih, wihq, 8,  192, 3 * HID);
    convert_swizzle<<<(32 * 192 * 512 + 255) / 256, 256, 0, stream>>>(whh, whhq, 32, 192, 3 * HID);
    convert_swizzle<<<(32 * 32  * 512 + 255) / 256, 256, 0, stream>>>(w3,  w3q,  32, 32,  HID / 2);

    float* out  = (float*)d_out;                 // [NENV, SEQ, 2]
    float* hout = out + (size_t)NENV * SEQ * 2;  // [NENV, HID]

    gru_decoder_kernel<<<NENV / ROWS, 256, 0, stream>>>(
        z, hxs, w1, b1, w2, b2, bih, bhh, b3, w4, b4,
        (const char*)wihq, (const char*)whhq, (const char*)w3q,
        out, hout);
}
